// SAKT_56968446214517
// MI455X (gfx1250) — compile-verified, matrix operand order checked by reading, not run on hardware
//
#include <hip/hip_runtime.h>
#include <math.h>

typedef __attribute__((ext_vector_type(16))) _Float16 v16h;
typedef __attribute__((ext_vector_type(8)))  _Float16 v8h;
typedef __attribute__((ext_vector_type(8)))  float    v8f;
typedef __attribute__((ext_vector_type(4)))  unsigned int v4u;
typedef __attribute__((ext_vector_type(8)))  int v8i;
typedef __attribute__((ext_vector_type(4)))  int v4i;

#if defined(__has_builtin)
#  if __has_builtin(__builtin_amdgcn_tensor_load_to_lds)
#    define USE_TDM 1
#  endif
#endif
#ifndef USE_TDM
#  define USE_TDM 0
#endif

namespace {
constexpr int kB = 16, kS = 512, kE = 256, kH = 4, kL = 2, kD = 64, kP = 10;
constexpr int kM = kB * kS;                 // 8192 rows
enum { MODE_QK = 0, MODE_VT = 1, MODE_RESID = 2 };
}

// ---- fragment helpers -------------------------------------------------------
// 16-bit A (16x32) / B(32x16, from transposed storage) fragment:
// lanes 0-15: elements 0..7 = row[k0+0..7],  8..15 = row[k0+16..23]
// lanes16-31: elements 0..7 = row[k0+8..15], 8..15 = row[k0+24..31]
__device__ __forceinline__ v16h load_frag(const _Float16* rowp, int k0, int hi) {
  v8h lo = *(const v8h*)(rowp + k0 + 8 * hi);
  v8h hg = *(const v8h*)(rowp + k0 + 16 + 8 * hi);
  v16h r;
#pragma unroll
  for (int i = 0; i < 8; ++i) { r[i] = lo[i]; r[i + 8] = hg[i]; }
  return r;
}

__device__ __forceinline__ v8f wmma16(v16h a, v16h b, v8f c) {
  return __builtin_amdgcn_wmma_f32_16x16x32_f16(false, a, false, b, (short)0, c,
                                                false, false);
}

__device__ __forceinline__ void wave_wait_lds() {
  asm volatile("s_wait_dscnt 0" ::: "memory");
}

#if USE_TDM
// Tensor Data Mover: DMA a [rows x cols] f32 tile (row stride = cols) from
// global memory into LDS at byte offset lds_off.  D# per CDNA5 ISA:
//  group0: [1:0]=count=1, [63:32]=lds_addr, [120:64]=global_addr, type=2
//  group1: data_size=4B, tensor_dim0=cols, tensor_dim1=rows,
//          tile_dim0=cols, tile_dim1=rows, dim0_stride=cols, dim1_stride=cols
__device__ __forceinline__ void tdm_load_2d_f32(const float* src,
                                                unsigned lds_off, int rows,
                                                int cols) {
  unsigned long long ga = (unsigned long long)src;
  v4u g0;
  g0[0] = 1u;                                            // count=1
  g0[1] = lds_off;                                       // lds_addr
  g0[2] = (unsigned)(ga & 0xffffffffu);                  // global_addr[31:0]
  g0[3] = (unsigned)((ga >> 32) & 0x01ffffffu) | (2u << 30);  // addr[56:32]|type
  v8i g1;
  g1[0] = (2 << 16);                  // workgroup_mask=0, data_size=2 (4 bytes)
  g1[1] = (cols & 0xffff) << 16;      // tensor_dim0[15:0]
  g1[2] = (rows & 0xffff) << 16;      // tensor_dim1[15:0]
  g1[3] = (cols & 0xffff) << 16;      // tile_dim0
  g1[4] = (rows & 0xffff);            // tile_dim1 (tile_dim2=0)
  g1[5] = cols;                       // tensor_dim0_stride[31:0]
  g1[6] = (cols & 0xffff) << 16;      // tensor_dim1_stride[15:0]
  g1[7] = 0;
  v4i z4 = {0, 0, 0, 0};
#if defined(__clang_major__) && __clang_major__ >= 23
  v8i z8 = {0, 0, 0, 0, 0, 0, 0, 0};
  __builtin_amdgcn_tensor_load_to_lds(g0, g1, z4, z4, z8, 0);
#else
  __builtin_amdgcn_tensor_load_to_lds(g0, g1, z4, z4, 0);
#endif
}
#endif

// ---- prep kernels -----------------------------------------------------------
__global__ void transpose_weights(const float* __restrict__ Wq,
                                  const float* __restrict__ Wk,
                                  const float* __restrict__ Wv,
                                  const float* __restrict__ Wo,
                                  _Float16* __restrict__ WqT,
                                  _Float16* __restrict__ WkT,
                                  _Float16* __restrict__ WvT,
                                  _Float16* __restrict__ WoT) {
  int idx = blockIdx.x * blockDim.x + threadIdx.x;
  if (idx >= kL * kE * kE) return;
  int l = idx / (kE * kE), rem = idx % (kE * kE);
  int n = rem / kE, k = rem % kE;
  int src = l * kE * kE + k * kE + n;      // WT[l][n][k] = W[l][k][n]
  WqT[idx] = (_Float16)Wq[src];
  WkT[idx] = (_Float16)Wk[src];
  WvT[idx] = (_Float16)Wv[src];
  WoT[idx] = (_Float16)Wo[src];
}

__global__ void gather_embed(const int* __restrict__ item_inputs,
                             const int* __restrict__ item_ids,
                             const float* __restrict__ input_embed,
                             const float* __restrict__ query_embed,
                             float* __restrict__ xf32,
                             _Float16* __restrict__ xf16,
                             _Float16* __restrict__ qbf16) {
  int idx = blockIdx.x * blockDim.x + threadIdx.x;
  if (idx >= kM * kE) return;
  int bs = idx / kE, e = idx % kE;
  float xv = input_embed[(long)item_inputs[bs] * kE + e];
  float qv = query_embed[(long)item_ids[bs] * kE + e];
  xf32[idx] = xv;
  xf16[idx] = (_Float16)xv;
  qbf16[idx] = (_Float16)qv;
}

// ---- projection GEMM: C[M=8192, N=256] = A @ W + b --------------------------
// block = 4 waves; wave -> 16(M) x 64(N) strip; K looped in 8 chunks of 32.
template <int MODE>
__global__ __launch_bounds__(128) void gemm_wmma(
    const _Float16* __restrict__ A,    // [M, 256] row-major f16
    const _Float16* __restrict__ WT,   // [256, 256] n-major (pre-transposed)
    const float* __restrict__ bias,    // [256]
    void* __restrict__ dst0, void* __restrict__ dst1) {
  const int lane = threadIdx.x & 31;
  const int wv = threadIdx.x >> 5;
  const int hi = lane >> 4, ln = lane & 15;
  const int m0 = blockIdx.x * 64 + wv * 16;
  const int n0 = blockIdx.y * 64;

  v8f acc[4];
#pragma unroll
  for (int t = 0; t < 4; ++t)
#pragma unroll
    for (int r = 0; r < 8; ++r) acc[t][r] = 0.f;

  const _Float16* arow = A + (m0 + ln) * kE;
  for (int kc = 0; kc < 8; ++kc) {
    int k0 = kc * 32;
    if (kc < 7) __builtin_prefetch(arow + k0 + 32, 0, 0);
    v16h af = load_frag(arow, k0, hi);
#pragma unroll
    for (int t = 0; t < 4; ++t) {
      const _Float16* brow = WT + (n0 + t * 16 + ln) * kE;
      v16h bf = load_frag(brow, k0, hi);
      acc[t] = wmma16(af, bf, acc[t]);
    }
  }

#pragma unroll
  for (int t = 0; t < 4; ++t) {
    int gn = n0 + t * 16 + ln;
    float bv = bias[gn];
#pragma unroll
    for (int r = 0; r < 8; ++r) {
      int gm = m0 + r + 8 * hi;
      float cv = acc[t][r] + bv;
      int bb = gm >> 9, ss = gm & 511, hh = gn >> 6, dd = gn & 63;
      if (MODE == MODE_QK) {          // [B,H,S,D] f16
        ((_Float16*)dst0)[(((bb * kH + hh) * kS) + ss) * kD + dd] = (_Float16)cv;
      } else if (MODE == MODE_VT) {   // [B,H,D,S] f16 (transposed V)
        ((_Float16*)dst0)[(((bb * kH + hh) * kD) + dd) * kS + ss] = (_Float16)cv;
      } else {                        // residual += relu(c); refresh f16 copy
        int idx = gm * kE + gn;
        float nv = ((float*)dst0)[idx] + fmaxf(cv, 0.f);
        ((float*)dst0)[idx] = nv;
        ((_Float16*)dst1)[idx] = (_Float16)nv;
      }
    }
  }
}

// ---- flash attention with relative-position terms ---------------------------
// one wave per (b,h,i-tile of 16 rows); score tiles via WMMA; online softmax.
// pos_key / pos_value tiles staged to LDS by the Tensor Data Mover; the 16x10
// q.pos_key table is computed in-kernel (rel=clip(i-j,0,9) has only 10 rows).
__global__ __launch_bounds__(128) void attn_flash(
    const _Float16* __restrict__ Qf,   // [B,H,S,D]
    const _Float16* __restrict__ Kf,   // [B,H,S,D]
    const _Float16* __restrict__ VT,   // [B,H,D,S]
    const float* __restrict__ pos_key,    // [10,64]
    const float* __restrict__ pos_value,  // [10,64]
    _Float16* __restrict__ merged) {      // [B,S,E]
  __shared__ float pklds[kP][kD];
  __shared__ float pvlds[kP][kD];
  __shared__ float qlds[4][16][kP];
  __shared__ float bucket[4][16][12];
  __shared__ __align__(16) _Float16 plds[4][16][32];

#if USE_TDM
  if (threadIdx.x == 0) {
    tdm_load_2d_f32(pos_key, (unsigned)(unsigned long long)(void*)&pklds[0][0],
                    kP, kD);
    tdm_load_2d_f32(pos_value, (unsigned)(unsigned long long)(void*)&pvlds[0][0],
                    kP, kD);
    __builtin_amdgcn_s_wait_tensorcnt(0);
  }
  __syncthreads();
#else
  for (int idx = threadIdx.x; idx < kP * kD; idx += 128) {
    pklds[idx / kD][idx % kD] = pos_key[idx];
    pvlds[idx / kD][idx % kD] = pos_value[idx];
  }
  __syncthreads();
#endif

  const int lane = threadIdx.x & 31;
  const int wv = threadIdx.x >> 5;
  const int hi = lane >> 4, ln = lane & 15;
  const int gw = blockIdx.x * 4 + wv;   // 0..2047
  const int it = gw & 31, bh = gw >> 5;
  const int b = bh >> 2, h = bh & 3;
  const int i0 = it * 16;
  const int jlimit = i0 + 16;           // exclusive causal j bound

  const _Float16* Qb = Qf + (size_t)bh * kS * kD;
  const _Float16* Kb = Kf + (size_t)bh * kS * kD;
  const _Float16* Vb = VT + (size_t)bh * kD * kS;

  // qpk[i-row][t] = Q[i0+row,:] . pos_key[t,:]  (5 dots per lane)
  for (int idx = lane; idx < 16 * kP; idx += 32) {
    int row = idx / kP, t = idx % kP;
    const _Float16* q = Qb + (i0 + row) * kD;
    float acc = 0.f;
#pragma unroll 8
    for (int d = 0; d < kD; ++d) acc += (float)q[d] * pklds[t][d];
    qlds[wv][row][t] = acc;
  }
  for (int idx = lane; idx < 16 * 12; idx += 32)
    bucket[wv][idx / 12][idx % 12] = 0.f;

  const _Float16* qr = Qb + (i0 + ln) * kD;
  v16h qf0 = load_frag(qr, 0, hi);
  v16h qf1 = load_frag(qr, 32, hi);

  float mrun[8], lrun[8];
  v8f o[4];
#pragma unroll
  for (int r = 0; r < 8; ++r) { mrun[r] = -INFINITY; lrun[r] = 0.f; }
#pragma unroll
  for (int t = 0; t < 4; ++t)
#pragma unroll
    for (int r = 0; r < 8; ++r) o[t][r] = 0.f;

  wave_wait_lds();
  float q9[8];
#pragma unroll
  for (int r = 0; r < 8; ++r) q9[r] = qlds[wv][r + 8 * hi][9];

  const float scale = 0.125f;  // 1/sqrt(64)
  const int nch = (jlimit + 31) >> 5;
  for (int c = 0; c < nch; ++c) {
    float s[2][8];
#pragma unroll
    for (int half2 = 0; half2 < 2; ++half2) {
      const int j0 = c * 32 + half2 * 16;
      if (j0 >= jlimit) {
#pragma unroll
        for (int r = 0; r < 8; ++r) s[half2][r] = -INFINITY;
        continue;
      }
      const _Float16* kr = Kb + (j0 + ln) * kD;
      v16h kf0 = load_frag(kr, 0, hi);
      v16h kf1 = load_frag(kr, 32, hi);
      v8f sc;
#pragma unroll
      for (int r = 0; r < 8; ++r) sc[r] = 0.f;
      sc = wmma16(qf0, kf0, sc);
      sc = wmma16(qf1, kf1, sc);
      const bool slow = (j0 + 24 > i0);  // tile touches rel<9 band / diagonal
      if (!slow) {
#pragma unroll
        for (int r = 0; r < 8; ++r) s[half2][r] = (sc[r] + q9[r]) * scale;
      } else {
#pragma unroll
        for (int r = 0; r < 8; ++r) {
          int m = r + 8 * hi;
          int d = i0 + m - (j0 + ln);     // i - j
          if (d < 0) s[half2][r] = -INFINITY;      // causal mask
          else {
            int t_ = d > 9 ? 9 : d;
            s[half2][r] = (sc[r] + qlds[wv][m][t_]) * scale;
          }
        }
      }
    }

    // online softmax update (row stats are uniform per (r, half))
    float mnew[8], scl[8];
#pragma unroll
    for (int r = 0; r < 8; ++r) {
      float v = fmaxf(s[0][r], s[1][r]);
      v = fmaxf(v, __shfl_xor(v, 1, 16));
      v = fmaxf(v, __shfl_xor(v, 2, 16));
      v = fmaxf(v, __shfl_xor(v, 4, 16));
      v = fmaxf(v, __shfl_xor(v, 8, 16));
      float mn = fmaxf(mrun[r], v);
      mnew[r] = mn;
      scl[r] = expf(mrun[r] - mn);
      mrun[r] = mn;
    }
#pragma unroll
    for (int t = 0; t < 4; ++t)
#pragma unroll
      for (int r = 0; r < 8; ++r) o[t][r] *= scl[r];
#pragma unroll
    for (int r = 0; r < 8; ++r) lrun[r] *= scl[r];
    if (ln < 10) {
#pragma unroll
      for (int r = 0; r < 8; ++r) bucket[wv][r + 8 * hi][ln] *= scl[r];
    }

    // P = exp(s - m); stage to LDS (A-layout source), bucket near-diagonal mass
#pragma unroll
    for (int half2 = 0; half2 < 2; ++half2) {
      const int j0 = c * 32 + half2 * 16;
      const bool diag = (j0 + 24 > i0) && (j0 < jlimit);
#pragma unroll
      for (int r = 0; r < 8; ++r) {
        int m = r + 8 * hi;
        float p = expf(s[half2][r] - mnew[r]);   // exp(-inf)=0 handles mask
        plds[wv][m][half2 * 16 + ln] = (_Float16)p;
        s[half2][r] = p;                          // reuse as prob
        if (diag) {
          int d = i0 + m - (j0 + ln);
          if (d >= 0 && d <= 8) atomicAdd(&bucket[wv][m][d], p);
        }
      }
    }
#pragma unroll
    for (int r = 0; r < 8; ++r) {
      float rs = s[0][r] + s[1][r];
      rs += __shfl_xor(rs, 1, 16);
      rs += __shfl_xor(rs, 2, 16);
      rs += __shfl_xor(rs, 4, 16);
      rs += __shfl_xor(rs, 8, 16);
      lrun[r] += rs;
    }

    wave_wait_lds();
    const int jb = c * 32;
    v16h pf = load_frag(&plds[wv][ln][0], 0, hi);
#pragma unroll
    for (int t = 0; t < 4; ++t) {
      const _Float16* vr = Vb + (t * 16 + ln) * kS + jb;
      v16h vf = load_frag(vr, 0, hi);
      o[t] = wmma16(pf, vf, o[t]);
    }
  }

  // finalize: O/l + pos_value[9] + sum_t (bucket_t/l)*(pos_value[t]-pos_value[9])
  wave_wait_lds();
#pragma unroll
  for (int r = 0; r < 8; ++r) {
    int m = r + 8 * hi;
    float inv = 1.f / lrun[r];
    float bu[9];
#pragma unroll
    for (int u = 0; u < 9; ++u) bu[u] = bucket[wv][m][u] * inv;
    _Float16* orow = merged + ((size_t)(b * kS + i0 + m) * kE + h * kD);
#pragma unroll
    for (int t = 0; t < 4; ++t) {
      int d = t * 16 + ln;
      float p9 = pvlds[9][d];
      float val = o[t][r] * inv + p9;
#pragma unroll
      for (int u = 0; u < 9; ++u) val += bu[u] * (pvlds[u][d] - p9);
      orow[d] = (_Float16)val;
    }
  }
}

// ---- final projection: out[m] = x[m,:] . out_w + out_b ----------------------
__global__ __launch_bounds__(256) void final_proj(const float* __restrict__ x,
                                                  const float* __restrict__ out_w,
                                                  const float* __restrict__ out_b,
                                                  float* __restrict__ out) {
  int row = (blockIdx.x * 256 + threadIdx.x) >> 5;
  int lane = threadIdx.x & 31;
  if (row >= kM) return;
  const float* xr = x + (size_t)row * kE;
  float acc = 0.f;
#pragma unroll
  for (int e = lane; e < kE; e += 32) acc += xr[e] * out_w[e];
  acc += __shfl_xor(acc, 16);
  acc += __shfl_xor(acc, 8);
  acc += __shfl_xor(acc, 4);
  acc += __shfl_xor(acc, 2);
  acc += __shfl_xor(acc, 1);
  if (lane == 0) out[row] = acc + out_b[0];
}

// ---- host -------------------------------------------------------------------
extern "C" void kernel_launch(void* const* d_in, const int* in_sizes, int n_in,
                              void* d_out, int out_size, void* d_ws,
                              size_t ws_size, hipStream_t stream) {
  (void)in_sizes; (void)n_in; (void)out_size; (void)ws_size;
  const int* item_inputs = (const int*)d_in[0];
  const int* item_ids = (const int*)d_in[1];
  const float* input_embed = (const float*)d_in[2];
  const float* query_embed = (const float*)d_in[3];
  const float* pos_key = (const float*)d_in[4];
  const float* pos_value = (const float*)d_in[5];
  const float* Wq = (const float*)d_in[6];
  const float* bq = (const float*)d_in[7];
  const float* Wk = (const float*)d_in[8];
  const float* bk = (const float*)d_in[9];
  const float* Wv = (const float*)d_in[10];
  const float* bv = (const float*)d_in[11];
  const float* Wo = (const float*)d_in[12];
  const float* bo = (const float*)d_in[13];
  const float* out_w = (const float*)d_in[14];
  const float* out_b = (const float*)d_in[15];
  float* out = (float*)d_out;

  char* p = (char*)d_ws;
  auto carve = [&](size_t bytes) -> char* {
    char* r = p;
    p += (bytes + 255) & ~(size_t)255;
    return r;
  };
  float* xf32 = (float*)carve((size_t)kM * kE * 4);
  _Float16* xf16 = (_Float16*)carve((size_t)kM * kE * 2);
  _Float16* qb16 = (_Float16*)carve((size_t)kM * kE * 2);
  _Float16* WqT = (_Float16*)carve((size_t)kL * kE * kE * 2);
  _Float16* WkT = (_Float16*)carve((size_t)kL * kE * kE * 2);
  _Float16* WvT = (_Float16*)carve((size_t)kL * kE * kE * 2);
  _Float16* WoT = (_Float16*)carve((size_t)kL * kE * kE * 2);
  _Float16* Qf = (_Float16*)carve((size_t)kB * kH * kS * kD * 2);
  _Float16* Kf = (_Float16*)carve((size_t)kB * kH * kS * kD * 2);
  _Float16* VTf = (_Float16*)carve((size_t)kB * kH * kD * kS * 2);
  _Float16* merged = (_Float16*)carve((size_t)kM * kE * 2);

  transpose_weights<<<(kL * kE * kE + 255) / 256, 256, 0, stream>>>(
      Wq, Wk, Wv, Wo, WqT, WkT, WvT, WoT);
  gather_embed<<<(kM * kE + 255) / 256, 256, 0, stream>>>(
      item_inputs, item_ids, input_embed, query_embed, xf32, xf16, qb16);

  dim3 ggrid(kM / 64, kE / 64);
  for (int l = 0; l < kL; ++l) {
    const size_t wof = (size_t)l * kE * kE;
    gemm_wmma<MODE_QK><<<ggrid, 128, 0, stream>>>(qb16, WqT + wof, bq + l * kE,
                                                  Qf, nullptr);
    gemm_wmma<MODE_QK><<<ggrid, 128, 0, stream>>>(xf16, WkT + wof, bk + l * kE,
                                                  Kf, nullptr);
    gemm_wmma<MODE_VT><<<ggrid, 128, 0, stream>>>(xf16, WvT + wof, bv + l * kE,
                                                  VTf, nullptr);
    attn_flash<<<512, 128, 0, stream>>>(Qf, Kf, VTf, pos_key, pos_value,
                                        merged);
    gemm_wmma<MODE_RESID><<<ggrid, 128, 0, stream>>>(merged, WoT + wof,
                                                     bo + l * kE, xf32, xf16);
  }
  final_proj<<<kM / 8, 256, 0, stream>>>(xf32, out_w, out_b, out);
}
